// Block_15461882266042
// MI455X (gfx1250) — matrix operand order repaired and checked by reading, compiled, not run on hardware
//
#include <hip/hip_runtime.h>
#include <hip/hip_bf16.h>

// ---------------- problem constants ----------------
#define BATCH      32
#define HH         56
#define WW_IMG     56
#define CDIM       256
#define NHEADS     8
#define HD         32
#define WS_        7
#define NTOK       49            // tokens per window
#define SHIFT_     3
#define NWIN       64            // windows per batch image (8x8)
#define TROWS      100352        // BATCH * 3136 = total window-tokens
#define ATT_SCALE  0.17677669529663687f  // 32^-0.5

typedef __attribute__((ext_vector_type(16))) __bf16 bf16x16;
typedef __attribute__((ext_vector_type(8)))  float  f32x8;

union Frag { unsigned int u[8]; bf16x16 v; };

__device__ __forceinline__ unsigned short f2b(float x) {
  unsigned int u = __float_as_uint(x);
  u += 0x7FFFu + ((u >> 16) & 1u);
  return (unsigned short)(u >> 16);
}
__device__ __forceinline__ unsigned int pack2(float a, float b) {
  return (unsigned int)f2b(a) | ((unsigned int)f2b(b) << 16);
}
__device__ __forceinline__ int regid(int x) { return x < 49 ? 0 : (x < 53 ? 1 : 2); }

// =====================================================================
// LayerNorm kernels. One wave (32 lanes) per 256-wide row.
// REMAP==1: LN1 path -- gather source row through shift + window partition
//           (output row index t is in window-token order).
// REMAP==0: LN2 path -- plain row order.
// =====================================================================
template<int REMAP>
__global__ __launch_bounds__(256)
void ln_kernel(const float* __restrict__ in, const float* __restrict__ g,
               const float* __restrict__ bta, unsigned short* __restrict__ out)
{
  const int t    = blockIdx.x * 8 + (threadIdx.x >> 5);
  const int lane = threadIdx.x & 31;

  const float* row;
  if (REMAP) {
    int w = t / NTOK, n = t - NTOK * w;
    int bb = w >> 6, wloc = w & 63;
    int wh = wloc >> 3, wwc = wloc & 7;
    int i = n / 7, j = n - 7 * i;
    int sh = wh * 7 + i + SHIFT_; if (sh >= HH) sh -= HH;
    int sw = wwc * 7 + j + SHIFT_; if (sw >= WW_IMG) sw -= WW_IMG;
    row = in + ((size_t)(bb * 3136 + sh * 56 + sw)) * CDIM;
  } else {
    row = in + (size_t)t * CDIM;
  }

  const int c = lane * 8;
  float4 v0 = *(const float4*)(row + c);
  float4 v1 = *(const float4*)(row + c + 4);
  float s = v0.x + v0.y + v0.z + v0.w + v1.x + v1.y + v1.z + v1.w;
  float q = v0.x*v0.x + v0.y*v0.y + v0.z*v0.z + v0.w*v0.w
          + v1.x*v1.x + v1.y*v1.y + v1.z*v1.z + v1.w*v1.w;
#pragma unroll
  for (int d = 16; d >= 1; d >>= 1) {
    s += __shfl_xor(s, d, 32);
    q += __shfl_xor(q, d, 32);
  }
  const float mu  = s * (1.0f / 256.0f);
  const float var = q * (1.0f / 256.0f) - mu * mu;
  const float inv = rsqrtf(var + 1e-5f);

  unsigned int o[4];
  o[0] = pack2((v0.x-mu)*inv*g[c+0]+bta[c+0], (v0.y-mu)*inv*g[c+1]+bta[c+1]);
  o[1] = pack2((v0.z-mu)*inv*g[c+2]+bta[c+2], (v0.w-mu)*inv*g[c+3]+bta[c+3]);
  o[2] = pack2((v1.x-mu)*inv*g[c+4]+bta[c+4], (v1.y-mu)*inv*g[c+5]+bta[c+5]);
  o[3] = pack2((v1.z-mu)*inv*g[c+6]+bta[c+6], (v1.w-mu)*inv*g[c+7]+bta[c+7]);
  *(uint4*)(out + (size_t)t * CDIM + c) = *(uint4*)o;
}

// =====================================================================
// bf16 GEMM with WMMA:  C[TROWS x Nfull] = A[TROWS x K](bf16) * B[K x Nfull](f32->bf16)
// Block tile 128x128, 8 waves; wave (wm,wn) owns rows 64*wm..+64, cols 32*wn..+32.
// K staged in 64-wide chunks.  EPI: 0=bias->bf16, 1=bias+GELU->bf16,
// 2=bias+window-reverse-scatter+residual(x)->f32, 3=bias+residual(x1)->f32.
// =====================================================================
template<int EPI>
__global__ __launch_bounds__(256)
void gemm_bf16(const unsigned short* __restrict__ A, const float* __restrict__ B,
               const float* __restrict__ bias, void* __restrict__ outv,
               const void* __restrict__ extrav, int K, int Nfull)
{
  __shared__ __align__(16) unsigned short As[128 * 66];   // row-major, pad 66
  __shared__ __align__(16) unsigned int   Bs[32 * 128];   // packed (k,k+1) pairs

  const int tid  = threadIdx.x;
  const int lane = tid & 31;
  const int wave = tid >> 5;
  const int wm = wave & 1, wn = wave >> 1;
  const int l15 = lane & 15, hsv = lane >> 4;
  const int rb = blockIdx.y * 128;
  const int nb = blockIdx.x * 128;

  f32x8 acc[4][2];
#pragma unroll
  for (int mt = 0; mt < 4; ++mt)
#pragma unroll
    for (int nt = 0; nt < 2; ++nt)
#pragma unroll
      for (int e = 0; e < 8; ++e) acc[mt][nt][e] = 0.0f;

  for (int kc = 0; kc < K; kc += 64) {
    __syncthreads();
    // ---- stage A chunk: 128 rows x 64 bf16 (copy, uint4 loads) ----
#pragma unroll
    for (int it = 0; it < 4; ++it) {
      int idx = tid + it * 256;                 // 0..1023
      int r = idx >> 3, c8 = (idx & 7) << 3;
      uint4 u = *(const uint4*)(A + (size_t)(rb + r) * K + kc + c8);
      unsigned short* dst = As + r * 66 + c8;
      *(unsigned int*)(dst + 0) = u.x;
      *(unsigned int*)(dst + 2) = u.y;
      *(unsigned int*)(dst + 4) = u.z;
      *(unsigned int*)(dst + 6) = u.w;
    }
    // ---- stage B chunk: 64 rows x 128 f32 -> packed bf16 pairs ----
#pragma unroll
    for (int it = 0; it < 16; ++it) {
      int idx = tid + it * 256;                 // 0..4095
      int k2 = idx >> 7, n = idx & 127;
      float b0 = B[(size_t)(kc + 2 * k2)     * Nfull + nb + n];
      float b1 = B[(size_t)(kc + 2 * k2 + 1) * Nfull + nb + n];
      Bs[idx] = pack2(b0, b1);
    }
    __syncthreads();

#pragma unroll
    for (int ks = 0; ks < 2; ++ks) {
      Frag a[4], b[2];
#pragma unroll
      for (int mt = 0; mt < 4; ++mt) {
        const unsigned short* rp = As + (64 * wm + 16 * mt + l15) * 66 + 32 * ks;
#pragma unroll
        for (int vv = 0; vv < 8; ++vv) {
          int kp = ((vv & 3) * 2) + 8 * hsv + 16 * (vv >> 2);
          a[mt].u[vv] = *(const unsigned int*)(rp + kp);
        }
      }
#pragma unroll
      for (int nt = 0; nt < 2; ++nt) {
#pragma unroll
        for (int vv = 0; vv < 8; ++vv)
          b[nt].u[vv] = Bs[(16 * ks + (vv & 3) + 4 * hsv + 8 * (vv >> 2)) * 128 + 32 * wn + 16 * nt + l15];
      }
#pragma unroll
      for (int mt = 0; mt < 4; ++mt)
#pragma unroll
        for (int nt = 0; nt < 2; ++nt)
          acc[mt][nt] = __builtin_amdgcn_wmma_f32_16x16x32_bf16(
              false, a[mt].v, false, b[nt].v, (short)0, acc[mt][nt], false, false);
    }
  }

  // ---- epilogue: r outer so per-row remap is computed once ----
#pragma unroll
  for (int mt = 0; mt < 4; ++mt) {
#pragma unroll
    for (int r = 0; r < 8; ++r) {
      const int row = rb + 64 * wm + 16 * mt + r + 8 * hsv;
      size_t posbase = 0;
      if (EPI == 2) {
        int w = row / NTOK, n = row - NTOK * w;
        int bb = w >> 6, wloc = w & 63;
        int wh = wloc >> 3, wwc = wloc & 7;
        int i = n / 7, j = n - 7 * i;
        int sh = wh * 7 + i + SHIFT_; if (sh >= HH) sh -= HH;
        int sw = wwc * 7 + j + SHIFT_; if (sw >= WW_IMG) sw -= WW_IMG;
        posbase = ((size_t)(bb * 3136 + sh * 56 + sw)) * CDIM;
      } else if (EPI == 3) {
        posbase = (size_t)row * CDIM;
      }
#pragma unroll
      for (int nt = 0; nt < 2; ++nt) {
        const int col = nb + 32 * wn + 16 * nt + l15;
        float v = acc[mt][nt][r] + bias[col];
        if (EPI == 0) {
          ((unsigned short*)outv)[(size_t)row * Nfull + col] = f2b(v);
        } else if (EPI == 1) {
          float t3 = 0.7978845608028654f * (v + 0.044715f * v * v * v);
          float gl = 0.5f * v * (1.0f + tanhf(t3));
          ((unsigned short*)outv)[(size_t)row * Nfull + col] = f2b(gl);
        } else {
          size_t pos = posbase + col;
          ((float*)outv)[pos] = ((const float*)extrav)[pos] + v;
        }
      }
    }
  }
}

// =====================================================================
// Windowed attention: block = 128 threads = 4 waves = 4 heads of one window.
// grid = (2048 windows, 2 head groups).  N=49 padded to 64.
// Softmax epilogue fully branchless: per-head bias row staged in LDS,
// column-side terms hoisted, validity applied as additive penalties so the
// bias LDS gathers are unconditional (no exec-mask regions).
// =====================================================================
__global__ __launch_bounds__(128)
void attn_kernel(const unsigned short* __restrict__ qkv,      // [TROWS][768] bf16
                 const float* __restrict__ bias_table,        // [169][8] f32
                 unsigned short* __restrict__ outp)           // [TROWS][256] bf16
{
  __shared__ __align__(16) unsigned int   Qs[4][64 * 17];     // hd-pairs per query row
  __shared__ __align__(16) unsigned int   Ks[4][64 * 17];     // hd-pairs per key row
  __shared__ __align__(16) unsigned short Vt[4][32 * 66];     // V transposed [hd][tok]
  __shared__ __align__(16) unsigned short Ph[4][16 * 66];     // P row-block scratch
  __shared__ __align__(16) float          Bt[4][172];         // per-head rel-pos bias row

  const int tid = threadIdx.x, lane = tid & 31, wid = tid >> 5;
  const int l15 = lane & 15, hsv = lane >> 4;
  const int win = blockIdx.x;
  const int h   = blockIdx.y * 4 + wid;
  const int bt  = win * NTOK;
  const int wloc = win & 63;
  const int wh = wloc >> 3, wwc = wloc & 7;

  unsigned int*   qs = Qs[wid];
  unsigned int*   ks = Ks[wid];
  unsigned short* vt = Vt[wid];
  unsigned short* ph = Ph[wid];
  float*          btl = Bt[wid];

  // ---- stage per-head bias row: Bt[idx] = bias_table[idx][h] ----
  for (int idx = lane; idx < 169; idx += 32)
    btl[idx] = bias_table[idx * NHEADS + h];

  // ---- stage Q, K (u32 hd-pairs), zero-pad rows >= 49 ----
  for (int idx = lane; idx < 64 * 16; idx += 32) {
    int r = idx >> 4, c2 = idx & 15;
    unsigned int qv = 0u, kv = 0u;
    if (r < NTOK) {
      const unsigned short* rp = qkv + (size_t)(bt + r) * 768 + h * 32 + 2 * c2;
      qv = *(const unsigned int*)(rp);
      kv = *(const unsigned int*)(rp + 256);
    }
    qs[r * 17 + c2] = qv;
    ks[r * 17 + c2] = kv;
  }
  // ---- stage V transposed ----
  for (int idx = lane; idx < 64 * 32; idx += 32) {
    int r = idx >> 5, hd = idx & 31;
    unsigned short vv = 0;
    if (r < NTOK) vv = qkv[(size_t)(bt + r) * 768 + 512 + h * 32 + hd];
    vt[hd * 66 + r] = vv;
  }
  asm volatile("s_wait_dscnt 0" ::: "memory");

  // ---- hoist K^T B-fragments (4 key tiles) and V B-fragments (2 hd x 2 ktok) ----
  Frag bk[4];
#pragma unroll
  for (int nt = 0; nt < 4; ++nt)
#pragma unroll
    for (int vv = 0; vv < 8; ++vv)
      bk[nt].u[vv] = ks[(16 * nt + l15) * 17 + (vv & 3) + 4 * hsv + 8 * (vv >> 2)];

  Frag bvv[2][2];
  const unsigned int* vtu = (const unsigned int*)vt;          // [32][33] u32 view
#pragma unroll
  for (int n2 = 0; n2 < 2; ++n2)
#pragma unroll
    for (int kt = 0; kt < 2; ++kt)
#pragma unroll
      for (int vv = 0; vv < 8; ++vv)
        bvv[n2][kt].u[vv] = vtu[(16 * n2 + l15) * 33 + (vv & 3) + 4 * hsv + 8 * (vv >> 2) + 16 * kt];

  // ---- hoist column-side terms (depend only on nt & lane) ----
  int   colterm[4];   // (6 - i2) + 13*(6 - j2)
  int   cid[4];       // column region id
  float cpen[4];      // 0 if key token valid, else -1e38
#pragma unroll
  for (int nt = 0; nt < 4; ++nt) {
    int cc = 16 * nt + l15;
    int i2 = cc / 7, j2 = cc - 7 * i2;
    colterm[nt] = (6 - i2) + 13 * (6 - j2);
    cid[nt]     = regid(wh * 7 + i2) * 3 + regid(wwc * 7 + j2);
    cpen[nt]    = (cc < NTOK) ? 0.0f : -1.0e38f;
  }

  // ---- per 16-query row-block ----
#pragma unroll 1
  for (int mt = 0; mt < 4; ++mt) {
    Frag aq;
#pragma unroll
    for (int vv = 0; vv < 8; ++vv)
      aq.u[vv] = qs[(16 * mt + l15) * 17 + (vv & 3) + 4 * hsv + 8 * (vv >> 2)];

    f32x8 s[4];
#pragma unroll
    for (int nt = 0; nt < 4; ++nt) {
#pragma unroll
      for (int e = 0; e < 8; ++e) s[nt][e] = 0.0f;
      s[nt] = __builtin_amdgcn_wmma_f32_16x16x32_bf16(
          false, aq.v, false, bk[nt].v, (short)0, s[nt], false, false);
    }

    // scale + rel-pos bias + shift mask + pad penalties, then row softmax
#pragma unroll
    for (int r = 0; r < 8; ++r) {
      int rr = 16 * mt + r + 8 * hsv;            // query token index (row)
      int i1 = rr / 7, j1 = rr - 7 * i1;
      int rowterm = i1 + 13 * j1;
      int rid = regid(wh * 7 + i1) * 3 + regid(wwc * 7 + j1);
      float rpen = (rr < NTOK) ? 0.0f : -1.0e38f;
      float vals[4];
      float m = -3.0e38f;
#pragma unroll
      for (int nt = 0; nt < 4; ++nt) {
        int bidx = rowterm + colterm[nt];
        bidx = bidx < 0 ? 0 : (bidx > 168 ? 168 : bidx);     // clamp, branchless
        float badd = btl[bidx];                               // unconditional LDS gather
        float pen  = (cid[nt] == rid) ? 0.0f : -100.0f;
        float v = fmaf(s[nt][r], ATT_SCALE, badd) + pen + cpen[nt] + rpen;
        vals[nt] = v;
        m = fmaxf(m, v);
      }
#pragma unroll
      for (int d = 8; d >= 1; d >>= 1) m = fmaxf(m, __shfl_xor(m, d, 32));
      float sum = 0.0f;
#pragma unroll
      for (int nt = 0; nt < 4; ++nt) { vals[nt] = __expf(vals[nt] - m); sum += vals[nt]; }
#pragma unroll
      for (int d = 8; d >= 1; d >>= 1) sum += __shfl_xor(sum, d, 32);
      float inv = 1.0f / sum;
      int prow = r + 8 * hsv;                    // row within this 16-row block
#pragma unroll
      for (int nt = 0; nt < 4; ++nt)
        ph[prow * 66 + 16 * nt + l15] = f2b(vals[nt] * inv);
    }
    asm volatile("s_wait_dscnt 0" ::: "memory");

    // P(16x64) * V(64x32)
    f32x8 o[2];
#pragma unroll
    for (int n2 = 0; n2 < 2; ++n2)
#pragma unroll
      for (int e = 0; e < 8; ++e) o[n2][e] = 0.0f;
#pragma unroll
    for (int kt = 0; kt < 2; ++kt) {
      Frag ap;
      const unsigned short* rp = ph + l15 * 66 + 32 * kt;
#pragma unroll
      for (int vv = 0; vv < 8; ++vv) {
        int kp = ((vv & 3) * 2) + 8 * hsv + 16 * (vv >> 2);
        ap.u[vv] = *(const unsigned int*)(rp + kp);
      }
#pragma unroll
      for (int n2 = 0; n2 < 2; ++n2)
        o[n2] = __builtin_amdgcn_wmma_f32_16x16x32_bf16(
            false, ap.v, false, bvv[n2][kt].v, (short)0, o[n2], false, false);
    }
    // store valid rows, head-major channel layout [t][h*32+hd]
#pragma unroll
    for (int n2 = 0; n2 < 2; ++n2)
#pragma unroll
      for (int r = 0; r < 8; ++r) {
        int rr = 16 * mt + r + 8 * hsv;
        if (rr < NTOK)
          outp[(size_t)(bt + rr) * CDIM + h * 32 + 16 * n2 + l15] = f2b(o[n2][r]);
      }
  }
}

// =====================================================================
// Launch. Workspace layout (needs ~392 MiB, 256B aligned sections):
//   xw    bf16 [TROWS*256]   (LN1 + shift + window-partition output)
//   qkvh  bf16 [TROWS*1024]  (qkv, later reused as MLP hidden)
//   attb  bf16 [TROWS*256]   (attn output, later reused as LN2 output)
//   x1    f32  [TROWS*256]   (residual after attention)
// =====================================================================
extern "C" void kernel_launch(void* const* d_in, const int* in_sizes, int n_in,
                              void* d_out, int out_size, void* d_ws, size_t ws_size,
                              hipStream_t stream) {
  const float* x      = (const float*)d_in[0];
  const float* n1g    = (const float*)d_in[1];
  const float* n1b    = (const float*)d_in[2];
  const float* qkv_w  = (const float*)d_in[3];
  const float* qkv_b  = (const float*)d_in[4];
  const float* proj_w = (const float*)d_in[5];
  const float* proj_b = (const float*)d_in[6];
  const float* btab   = (const float*)d_in[7];
  const float* n2g    = (const float*)d_in[8];
  const float* n2b    = (const float*)d_in[9];
  const float* w1     = (const float*)d_in[10];
  const float* b1     = (const float*)d_in[11];
  const float* w2     = (const float*)d_in[12];
  const float* b2     = (const float*)d_in[13];
  float* out = (float*)d_out;

  char* ws = (char*)d_ws;
  const size_t SZ_XW   = (size_t)TROWS * 256 * 2;   // 51,380,224
  const size_t SZ_QKVH = (size_t)TROWS * 1024 * 2;  // 205,520,896
  const size_t SZ_ATTB = (size_t)TROWS * 256 * 2;
  unsigned short* xw   = (unsigned short*)(ws);
  unsigned short* qkvh = (unsigned short*)(ws + SZ_XW);
  unsigned short* attb = (unsigned short*)(ws + SZ_XW + SZ_QKVH);
  float*          x1   = (float*)(ws + SZ_XW + SZ_QKVH + SZ_ATTB);

  const int LN_BLOCKS = TROWS / 8;            // 12544
  const int MT = TROWS / 128;                 // 784 M-tiles

  // 1. LN1 + shift + window partition -> xw (bf16, window-token order)
  ln_kernel<1><<<LN_BLOCKS, 256, 0, stream>>>(x, n1g, n1b, xw);
  // 2. QKV GEMM: xw @ qkv_w + qkv_b -> qkvh (bf16 [TROWS][768])
  gemm_bf16<0><<<dim3(768 / 128, MT), 256, 0, stream>>>(xw, qkv_w, qkv_b, qkvh, nullptr, 256, 768);
  // 3. Windowed attention (bias + mask + softmax) -> attb (bf16 [TROWS][256])
  attn_kernel<<<dim3(TROWS / NTOK, 2), 128, 0, stream>>>(qkvh, btab, attb);
  // 4. Proj GEMM + window reverse + un-shift + residual(x) -> x1 (f32)
  gemm_bf16<2><<<dim3(256 / 128, MT), 256, 0, stream>>>(attb, proj_w, proj_b, x1, x, 256, 256);
  // 5. LN2 -> attb reused (bf16)
  ln_kernel<0><<<LN_BLOCKS, 256, 0, stream>>>(x1, n2g, n2b, attb);
  // 6. MLP1 + GELU -> qkvh reused (bf16 [TROWS][1024])
  gemm_bf16<1><<<dim3(1024 / 128, MT), 256, 0, stream>>>(attb, w1, b1, qkvh, nullptr, 256, 1024);
  // 7. MLP2 + residual(x1) -> d_out (f32)
  gemm_bf16<3><<<dim3(256 / 128, MT), 256, 0, stream>>>(qkvh, w2, b2, out, x1, 1024, 256);
}
